// Attention_87454124081847
// MI455X (gfx1250) — matrix operand order repaired and checked by reading, not compiled, not run on hardware
//
#include <hip/hip_runtime.h>
#include <hip/hip_bf16.h>
#include <math.h>

// Problem constants (fixed by the reference)
#define Bn  4
#define Sn  2048
#define Dn  1024
#define Hn  16
#define HDn 64
#define Mn  (Bn * Sn)   // 8192 total rows of x

typedef __attribute__((ext_vector_type(16))) __bf16 v16bf;
typedef __attribute__((ext_vector_type(8)))  float  v8f;

union Frag {
    v16bf v;
    uint4 q[2];
    unsigned short us[16];
};

// ---------------------------------------------------------------------------
// CDNA5 async global->LDS copy (ASYNCcnt-tracked), with sync fallback.
// Builtin signature (probe-confirmed by round-2 diagnostic):
//   void __builtin_amdgcn_global_load_async_to_lds_b128(
//       int4 addrspace(1)* gsrc, int4 addrspace(3)* ldst, imm int off, imm int cpol)
#if __has_builtin(__builtin_amdgcn_global_load_async_to_lds_b128)
#define HAVE_ASYNC_LDS 1
#else
#define HAVE_ASYNC_LDS 0
#endif

typedef int v4i_g __attribute__((vector_size(16)));
typedef __attribute__((address_space(1))) v4i_g* as1_v4i_p;
typedef __attribute__((address_space(3))) v4i_g* as3_v4i_p;

__device__ inline void async_cp16(const __bf16* g, __bf16* l) {
#if HAVE_ASYNC_LDS
    __builtin_amdgcn_global_load_async_to_lds_b128(
        (as1_v4i_p)g, (as3_v4i_p)l, 0, 0);
#else
    *(uint4*)l = *(const uint4*)g;    // global_load_b128 + ds_store_b128
#endif
}

#if HAVE_ASYNC_LDS
#if __has_builtin(__builtin_amdgcn_s_wait_asynccnt)
#define WAIT_ASYNC(n) __builtin_amdgcn_s_wait_asynccnt(n)
#else
#define WAIT_ASYNC(n) asm volatile("s_wait_asynccnt " #n ::: "memory")
#endif
#else
#define WAIT_ASYNC(n) ((void)0)
#endif

// ---------------------------------------------------------------------------
__device__ inline unsigned short f2bf_bits(float f) {
    union { float f; unsigned u; } x; x.f = f;
    unsigned r = x.u + 0x7FFFu + ((x.u >> 16) & 1u);  // round-to-nearest-even
    return (unsigned short)(r >> 16);
}
__device__ inline __bf16 f2bf(float f) {
    union { unsigned short s; __bf16 b; } o; o.s = f2bf_bits(f);
    return o.b;
}

__device__ inline v8f wmma_bf16(v16bf a, v16bf b, v8f c) {
    // D = A(16x32 bf16) * B(32x16 bf16) + C(16x16 f32)
    return __builtin_amdgcn_wmma_f32_16x16x32_bf16(
        false, a, false, b, (short)0, c, false, false);
}

// A-fragment (16x32 bf16, row-major with row stride ldr):
// lanes 0-15: M=lane, K = 0..15 ; lanes 16-31: M=lane-16, K = 16..31
__device__ inline v16bf load_a_frag(const __bf16* p, int ldr) {
    const int lane = threadIdx.x & 31;
    const int m = lane & 15;
    const int hh = (lane >> 4) * 16;
    Frag f;
    const __bf16* r = p + m * ldr;
    f.q[0] = *(const uint4*)(r + hh);
    f.q[1] = *(const uint4*)(r + hh + 8);
    return f.v;
}

// B-fragment (32x16 bf16). Memory holds B COLUMNS contiguously (row n of length-K
// storage == column n of B), row stride ldr:
// lanes 0-15: N=lane, K=0..15 ; lanes 16-31: N=lane-16, K=16..31
__device__ inline v16bf load_b_frag(const __bf16* p, int ldr) {
    const int lane = threadIdx.x & 31;
    const int n = lane & 15;
    const int hh = (lane >> 4) * 16;
    Frag f;
    const __bf16* r = p + n * ldr + hh;
    f.q[0] = *(const uint4*)(r);
    f.q[1] = *(const uint4*)(r + 8);
    return f.v;
}

// ---------------------------------------------------------------------------
__global__ void cvt_f32_to_bf16(const float* __restrict__ in,
                                __bf16* __restrict__ out, int n) {
    int i = blockIdx.x * blockDim.x + threadIdx.x;
    int stride = gridDim.x * blockDim.x;
    for (; i < n; i += stride) out[i] = f2bf(in[i]);
}

// ---------------------------------------------------------------------------
// C[M,N] = A[M,K] * W[N,K]^T  (y = x @ W.T). Block: 256 threads (8 waves),
// block tile 128M x 128N, wave tile 32M x 64N (waves 4 rows x 2 cols).
// Double-buffered LDS staging via async global->LDS copies.
// mode 0: store bf16 head-major [B,H,S,HD]          (Q,K projections)
// mode 1: store fp32 row-major  [M,N]               (output projection)
// mode 2: store bf16 head-TRANSPOSED [B,H,HD,S]     (V projection -> V^T)
__global__ __launch_bounds__(256)
void gemm_bf16(const __bf16* __restrict__ A, const __bf16* __restrict__ W,
               __bf16* __restrict__ outB, float* __restrict__ outF32, int mode) {
    __shared__ __align__(16) __bf16 lA[2][128 * 32];
    __shared__ __align__(16) __bf16 lB[2][128 * 32];

    const int tid  = threadIdx.x;
    const int w    = tid >> 5;
    const int lane = tid & 31;
    const int half = lane >> 4;
    const int nl   = lane & 15;
    const int wrow = w >> 1;          // 0..3 -> M offset 32*wrow
    const int wcol = w & 1;           // 0..1 -> N offset 64*wcol
    const int m0   = blockIdx.x * 128;
    const int n0   = blockIdx.y * 128;

    auto stage = [&](int buf, int kt) {
        #pragma unroll
        for (int t = 0; t < 2; ++t) {
            int v   = tid * 2 + t;        // 0..511
            int row = v >> 2;             // 0..127
            int c   = (v & 3) * 8;        // 0,8,16,24
            async_cp16(A + (size_t)(m0 + row) * Dn + kt + c, &lA[buf][row * 32 + c]);
            async_cp16(W + (size_t)(n0 + row) * Dn + kt + c, &lB[buf][row * 32 + c]);
        }
    };

    v8f acc[2][4] = {};

    stage(0, 0);
    const int NK = Dn / 32;
    for (int it = 0; it < NK; ++it) {
        const int cur = it & 1;
        if (it + 1 < NK) {
            stage(cur ^ 1, (it + 1) * 32);   // prefetch next tile into other buffer
            WAIT_ASYNC(4);                   // ensure current tile's 4 issues done
        } else {
            WAIT_ASYNC(0);
        }
        __syncthreads();

        v16bf a0 = load_a_frag(&lA[cur][(wrow * 32) * 32], 32);
        v16bf a1 = load_a_frag(&lA[cur][(wrow * 32 + 16) * 32], 32);
        #pragma unroll
        for (int j = 0; j < 4; ++j) {
            v16bf b = load_b_frag(&lB[cur][(wcol * 64 + j * 16) * 32], 32);
            acc[0][j] = wmma_bf16(a0, b, acc[0][j]);
            acc[1][j] = wmma_bf16(a1, b, acc[1][j]);
        }
        __syncthreads();
    }

    #pragma unroll
    for (int r = 0; r < 2; ++r) {
        #pragma unroll
        for (int j = 0; j < 4; ++j) {
            #pragma unroll
            for (int i = 0; i < 8; ++i) {
                int gm = m0 + wrow * 32 + r * 16 + i + half * 8;  // C: VGPR i -> row i+8*half
                int gn = n0 + wcol * 64 + j * 16 + nl;            // lane%16 -> column
                float val = acc[r][j][i];
                if (mode == 1) {
                    outF32[(size_t)gm * Dn + gn] = val;
                } else {
                    int bb = gm / Sn, ss = gm % Sn;
                    int hh = gn / HDn, dd = gn % HDn;
                    size_t base = ((size_t)bb * Hn + hh);
                    if (mode == 0)
                        outB[(base * Sn + ss) * HDn + dd] = f2bf(val);
                    else  // mode 2: transposed-in-head [B,H,HD,S]
                        outB[(base * HDn + dd) * Sn + ss] = f2bf(val);
                }
            }
        }
    }
}

// ---------------------------------------------------------------------------
// Flash attention. Q,K head-major [B,H,S,HD]; V pre-transposed [B,H,HD,S].
// Block = (qblock of 64, head, batch), 128 threads (4 waves x 16 queries),
// 32 keys per iteration, double-buffered async K/V^T staging.
__global__ __launch_bounds__(128)
void flash_attn(const __bf16* __restrict__ Qh, const __bf16* __restrict__ Kh,
                const __bf16* __restrict__ Vt, float* __restrict__ perHead,
                __bf16* __restrict__ comb) {
    __shared__ __align__(16) __bf16 lK[2][32 * 64];     // keys: 32 rows x 64 d
    __shared__ __align__(16) __bf16 lVt[2][64 * 32];    // V^T : 64 d   x 32 keys
    __shared__ __align__(16) __bf16 lP[4][16 * 32];     // per-wave P tile (16q x 32k)

    const int tid  = threadIdx.x;
    const int w    = tid >> 5;
    const int lane = tid & 31;
    const int half = lane >> 4;
    const int nl   = lane & 15;
    const int hh   = blockIdx.y;
    const int bb   = blockIdx.z;

    const size_t headBase = ((size_t)bb * Hn + hh) * (size_t)Sn * HDn;  // same for Kh & Vt
    const int q0 = blockIdx.x * 64 + w * 16;

    v16bf qa0 = load_a_frag(Qh + headBase + (size_t)q0 * HDn, HDn);       // d 0..31
    v16bf qa1 = load_a_frag(Qh + headBase + (size_t)q0 * HDn + 32, HDn);  // d 32..63

    auto stage = [&](int buf, int kt) {
        #pragma unroll
        for (int t = 0; t < 2; ++t) {
            int v = tid * 2 + t;          // 0..255
            // K tile: row = key (128B contiguous per row)
            int kr = v >> 3, kc = (v & 7) * 8;
            async_cp16(Kh + headBase + (size_t)(kt + kr) * HDn + kc,
                       &lK[buf][kr * 64 + kc]);
            // V^T tile: row = d, 32 contiguous keys (64B per row, stride S)
            int vr = v >> 2, vc = (v & 3) * 8;
            async_cp16(Vt + headBase + (size_t)vr * Sn + kt + vc,
                       &lVt[buf][vr * 32 + vc]);
        }
    };

    v8f o0 = {}, o1 = {}, o2 = {}, o3 = {};
    float mrow[8], lrow[8];
    #pragma unroll
    for (int i = 0; i < 8; ++i) { mrow[i] = -1e30f; lrow[i] = 0.f; }

    const float scale = 0.125f;   // 1/sqrt(HD)
    const int NT = Sn / 32;

    stage(0, 0);
    for (int it = 0; it < NT; ++it) {
        const int cur = it & 1;
        if (it + 1 < NT) {
            stage(cur ^ 1, (it + 1) * 32);
            WAIT_ASYNC(4);
        } else {
            WAIT_ASYNC(0);
        }
        __syncthreads();

        // scores: S(16q x 32k) = Q(16x64) * K^T ; B columns = key rows (contiguous)
        v8f s0 = {}, s1 = {};
        s0 = wmma_bf16(qa0, load_b_frag(&lK[cur][0], 64), s0);
        s0 = wmma_bf16(qa1, load_b_frag(&lK[cur][32], 64), s0);
        s1 = wmma_bf16(qa0, load_b_frag(&lK[cur][16 * 64], 64), s1);
        s1 = wmma_bf16(qa1, load_b_frag(&lK[cur][16 * 64 + 32], 64), s1);

        // online softmax; row (query) = i + 8*half, spread across 16 lanes
        #pragma unroll
        for (int i = 0; i < 8; ++i) {
            float a0 = s0[i] * scale, a1 = s1[i] * scale;
            float rmax = fmaxf(a0, a1);
            #pragma unroll
            for (int mask = 8; mask >= 1; mask >>= 1)
                rmax = fmaxf(rmax, __shfl_xor(rmax, mask, 32));
            float mnew = fmaxf(mrow[i], rmax);
            float corr = __expf(mrow[i] - mnew);
            float p0 = __expf(a0 - mnew);
            float p1 = __expf(a1 - mnew);
            float rsum = p0 + p1;
            #pragma unroll
            for (int mask = 8; mask >= 1; mask >>= 1)
                rsum += __shfl_xor(rsum, mask, 32);
            lrow[i] = lrow[i] * corr + rsum;
            mrow[i] = mnew;
            o0[i] *= corr; o1[i] *= corr; o2[i] *= corr; o3[i] *= corr;
            // park P in LDS so it can be re-read in A-fragment layout
            lP[w][(i + 8 * half) * 32 + nl]      = f2bf(p0);
            lP[w][(i + 8 * half) * 32 + 16 + nl] = f2bf(p1);
        }
        // intra-wave LDS RAW: wait for ds stores before fragment reload
        asm volatile("s_wait_dscnt 0" ::: "memory");

        // O(16q x 64d) += P(16x32) * V(32x64); lVt rows = d, contiguous over keys
        v16bf pa = load_a_frag(&lP[w][0], 32);
        o0 = wmma_bf16(pa, load_b_frag(&lVt[cur][0], 32), o0);
        o1 = wmma_bf16(pa, load_b_frag(&lVt[cur][16 * 32], 32), o1);
        o2 = wmma_bf16(pa, load_b_frag(&lVt[cur][32 * 32], 32), o2);
        o3 = wmma_bf16(pa, load_b_frag(&lVt[cur][48 * 32], 32), o3);
        __syncthreads();
    }

    // epilogue: normalize, write per_head fp32 [B,S,H,HD] + bf16 combined copy
    #pragma unroll
    for (int i = 0; i < 8; ++i) {
        int q = q0 + i + 8 * half;
        float inv = 1.0f / lrow[i];
        float vals[4] = { o0[i] * inv, o1[i] * inv, o2[i] * inv, o3[i] * inv };
        #pragma unroll
        for (int dj = 0; dj < 4; ++dj) {
            int dd = dj * 16 + nl;
            size_t idx = (((size_t)bb * Sn + q) * Hn + hh) * HDn + dd;
            perHead[idx] = vals[dj];
            comb[idx] = f2bf(vals[dj]);
        }
    }
}

// ---------------------------------------------------------------------------
extern "C" void kernel_launch(void* const* d_in, const int* in_sizes, int n_in,
                              void* d_out, int out_size, void* d_ws, size_t ws_size,
                              hipStream_t stream) {
    const float* x  = (const float*)d_in[0];
    const float* Wq = (const float*)d_in[1];
    const float* Wk = (const float*)d_in[2];
    const float* Wv = (const float*)d_in[3];
    const float* Wo = (const float*)d_in[4];

    char* ws = (char*)d_ws;
    size_t off = 0;
    auto carve = [&](size_t bytes) -> void* {
        void* p = ws + off;
        off += (bytes + 255) & ~(size_t)255;
        return p;
    };
    const size_t nx = (size_t)Mn * Dn;   // 8.4M elements
    const size_t nw = (size_t)Dn * Dn;   // 1.05M elements

    __bf16* xb   = (__bf16*)carve(nx * 2);
    __bf16* wqb  = (__bf16*)carve(nw * 2);
    __bf16* wkb  = (__bf16*)carve(nw * 2);
    __bf16* wvb  = (__bf16*)carve(nw * 2);
    __bf16* wob  = (__bf16*)carve(nw * 2);
    __bf16* Qh   = (__bf16*)carve(nx * 2);
    __bf16* Kh   = (__bf16*)carve(nx * 2);
    __bf16* Vt   = (__bf16*)carve(nx * 2);   // [B,H,HD,S]
    __bf16* comb = (__bf16*)carve(nx * 2);

    float* projected = (float*)d_out;            // output #1: [B,S,D]
    float* perHead   = projected + nx;           // output #2: [B,S,H,HD]

    // 1) fp32 -> bf16 staging
    cvt_f32_to_bf16<<<(int)((nx + 255) / 256), 256, 0, stream>>>(x, xb, (int)nx);
    cvt_f32_to_bf16<<<(int)((nw + 255) / 256), 256, 0, stream>>>(Wq, wqb, (int)nw);
    cvt_f32_to_bf16<<<(int)((nw + 255) / 256), 256, 0, stream>>>(Wk, wkb, (int)nw);
    cvt_f32_to_bf16<<<(int)((nw + 255) / 256), 256, 0, stream>>>(Wv, wvb, (int)nw);
    cvt_f32_to_bf16<<<(int)((nw + 255) / 256), 256, 0, stream>>>(Wo, wob, (int)nw);

    // 2) QKV projections (V stored pre-transposed per head)
    dim3 gg(Mn / 128, Dn / 128);
    gemm_bf16<<<gg, 256, 0, stream>>>(xb, wqb, Qh, nullptr, 0);
    gemm_bf16<<<gg, 256, 0, stream>>>(xb, wkb, Kh, nullptr, 0);
    gemm_bf16<<<gg, 256, 0, stream>>>(xb, wvb, Vt, nullptr, 2);

    // 3) attention with online softmax
    flash_attn<<<dim3(Sn / 64, Hn, Bn), 128, 0, stream>>>(Qh, Kh, Vt, perHead, comb);

    // 4) output projection -> fp32
    gemm_bf16<<<gg, 256, 0, stream>>>(comb, wob, nullptr, projected, 1);
}
